// AssociativeLayerWrapper_25907242729690
// MI455X (gfx1250) — matrix operand order repaired and checked by reading, not compiled, not run on hardware
//
#include <hip/hip_runtime.h>
#include <hip/hip_bf16.h>
#include <math.h>

// ---------------------------------------------------------------------------
// MI455X (gfx1250, wave32) implementation of the AssociativeLayer reference.
// All GEMMs use V_WMMA_F32_16X16X32_BF16 (bf16 operands, f32 accumulate).
// Main GEMM stages its operand slabs into LDS via async global->LDS DMA.
// ---------------------------------------------------------------------------

typedef __bf16 bf16_t;
typedef __attribute__((ext_vector_type(16))) __bf16 v16bf;
typedef __attribute__((ext_vector_type(8)))  __bf16 v8bf;
typedef __attribute__((ext_vector_type(8)))  float  v8f;
typedef __attribute__((ext_vector_type(8)))  float  f32x8;

// async global->LDS builtin takes int4* in AS1 (global) / AS3 (LDS)
typedef __attribute__((ext_vector_type(4))) int int4v;
typedef __attribute__((address_space(1))) int4v g_int4;
typedef __attribute__((address_space(3))) int4v l_int4;

#define EPSV 1e-5f

#define WMMA_BF16(a, b, c) \
  __builtin_amdgcn_wmma_f32_16x16x32_bf16(false, (a), false, (b), (short)0, (c), false, false)

#if defined(__gfx1250__) && __has_builtin(__builtin_amdgcn_global_load_async_to_lds_b128)
#define HAVE_ASYNC_LDS 1
#else
#define HAVE_ASYNC_LDS 0
#endif

static __device__ __forceinline__ void wait_async_lds() {
#if __has_builtin(__builtin_amdgcn_s_wait_asynccnt)
  __builtin_amdgcn_s_wait_asynccnt(0);
#else
  asm volatile("s_wait_asynccnt 0x0" ::: "memory");
#endif
}

// A fragment (16x32 MxK, bf16) from row-major [M][K] source.
// ISA layout: lanes 0-15 row M=lane, K = {0..7, 16..23}; lanes 16-31 row M=lane-16, K shifted +8.
static __device__ __forceinline__ v16bf load_a_bf16(const bf16_t* base, int ld, int m0, int k0) {
  const int lane = threadIdx.x & 31;
  const int half = lane >> 4;
  const int row  = m0 + (lane & 15);
  const bf16_t* p = base + (size_t)row * ld + k0 + half * 8;
  v8bf lo = *(const v8bf*)p;          // K = k0+half*8 .. +7
  v8bf hi = *(const v8bf*)(p + 16);   // K = k0+16+half*8 .. +7
  v16bf r;
#pragma unroll
  for (int i = 0; i < 8; ++i) { r[i] = lo[i]; r[i + 8] = hi[i]; }
  return r;
}

// Same A fragment but from fp32 source, converted in-register to bf16.
static __device__ __forceinline__ v16bf load_a_f32(const float* base, int ld, int m0, int k0) {
  const int lane = threadIdx.x & 31;
  const int half = lane >> 4;
  const int row  = m0 + (lane & 15);
  const float* p = base + (size_t)row * ld + k0 + half * 8;
  f32x8 lo = *(const f32x8*)p;
  f32x8 hi = *(const f32x8*)(p + 16);
  v16bf r;
#pragma unroll
  for (int i = 0; i < 8; ++i) { r[i] = (bf16_t)lo[i]; r[i + 8] = (bf16_t)hi[i]; }
  return r;
}

// B fragment (32x16 KxN, bf16) from an [N][K]-major source (per-col contiguous K).
// ISA layout: lanes 0-15 col N=lane, K = {0..7, 16..23}; lanes 16-31 col N=lane-16, K shifted +8.
static __device__ __forceinline__ v16bf load_b_bf16(const bf16_t* base, int ld, int n0, int k0) {
  const int lane = threadIdx.x & 31;
  const int col  = n0 + (lane & 15);
  const int kk   = k0 + (lane >> 4) * 8;
  const bf16_t* p = base + (size_t)col * ld + kk;
  v8bf lo = *(const v8bf*)p;
  v8bf hi = *(const v8bf*)(p + 16);
  v16bf r;
#pragma unroll
  for (int i = 0; i < 8; ++i) { r[i] = lo[i]; r[i + 8] = hi[i]; }
  return r;
}

// ---------------------------------------------------------------------------
// K0: fp32 -> bf16 weight conversion (+ transpose for W_mem)
// ---------------------------------------------------------------------------
__global__ __launch_bounds__(256) void cvt_kernel(const float* __restrict__ src,
                                                  bf16_t* __restrict__ dst, int n) {
  int i = blockIdx.x * 256 + threadIdx.x;
  if (i < n) dst[i] = (bf16_t)src[i];
}

// W_mem [384][2048] fp32 -> WmemT [2048][384] bf16
__global__ __launch_bounds__(256) void transpose_cvt_kernel(const float* __restrict__ W,
                                                            bf16_t* __restrict__ WT) {
  int i = blockIdx.x * 256 + threadIdx.x;
  if (i < 384 * 2048) {
    int j = i / 2048, t = i % 2048;
    WT[(size_t)t * 384 + j] = (bf16_t)W[i];
  }
}

// ---------------------------------------------------------------------------
// K1: q = Hrows @ Wq^T (rows x 64, K=2048) via WMMA, then fused dpfp:
//     mq[j] = x[j%128] * x[(j%128 - (j/128+1)) mod 128],  x = [relu(q),relu(-q)]
//     dot_z[row] = sum_j z[j]*mq[j];  sumsq[row] = sum_j mq[j]^2 (optional)
// Block = 256 thr (8 waves), 64 rows/block. Waves: mtile=w&3, npair=w>>2.
// ---------------------------------------------------------------------------
__global__ __launch_bounds__(256) void mq_kernel(
    const float*  __restrict__ Hrows,   // [nrows][2048]
    const bf16_t* __restrict__ Wq,      // [64][2048] bf16 (N-major, K contiguous)
    const float*  __restrict__ z,       // [384]
    bf16_t* __restrict__ mq_bf,         // [nrows][384]
    bf16_t* __restrict__ mqT_bf,        // [384][tokN] or null
    float*  __restrict__ mq_f32,        // [nrows][384] or null
    float*  __restrict__ dot_z,         // [nrows]
    float*  __restrict__ sumsq,         // [nrows] or null
    int tokN) {
  __shared__ float q[64][65];
  __shared__ float partz[256];
  __shared__ float partq[256];
  __shared__ float zsh[384];

  for (int i = threadIdx.x; i < 384; i += 256) zsh[i] = z[i];

  const int rowbase = blockIdx.x * 64;
  const int wave  = threadIdx.x >> 5;
  const int mtile = wave & 3;     // 0..3 -> 16-row strip
  const int npair = wave >> 2;    // 0..1 -> 32-col strip
  const float* A = Hrows + (size_t)rowbase * 2048;

  v8f c0 = {}; v8f c1 = {};
#pragma unroll 4
  for (int k0 = 0; k0 < 2048; k0 += 32) {
    v16bf a  = load_a_f32(A, 2048, mtile * 16, k0);
    v16bf b0 = load_b_bf16(Wq, 2048, npair * 32,      k0);
    v16bf b1 = load_b_bf16(Wq, 2048, npair * 32 + 16, k0);
    c0 = WMMA_BF16(a, b0, c0);
    c1 = WMMA_BF16(a, b1, c1);
  }
  {
    const int lane = threadIdx.x & 31;
    const int rr = mtile * 16 + (lane >> 4) * 8;
    const int cc = npair * 32 + (lane & 15);
#pragma unroll
    for (int v = 0; v < 8; ++v) {
      q[rr + v][cc]      = c0[v];
      q[rr + v][cc + 16] = c1[v];
    }
  }
  __syncthreads();

  // dpfp: 4 threads per row, 96 j's each
  const int r    = threadIdx.x >> 2;   // 0..63
  const int sub  = threadIdx.x & 3;    // 0..3
  const int grow = rowbase + r;
  float az = 0.f, aq = 0.f;
  for (int j = sub * 96; j < sub * 96 + 96; ++j) {
    const int blk = j >> 7;
    const int i   = j & 127;
    const int i2  = (i - (blk + 1)) & 127;
    float x1 = (i  < 64) ? fmaxf(q[r][i], 0.f)  : fmaxf(-q[r][i  - 64], 0.f);
    float x2 = (i2 < 64) ? fmaxf(q[r][i2], 0.f) : fmaxf(-q[r][i2 - 64], 0.f);
    float m = x1 * x2;
    az += zsh[j] * m;
    aq += m * m;
    mq_bf[(size_t)grow * 384 + j] = (bf16_t)m;
    if (mq_f32) mq_f32[(size_t)grow * 384 + j] = m;
    if (mqT_bf) mqT_bf[(size_t)j * tokN + grow] = (bf16_t)m;
  }
  partz[threadIdx.x] = az;
  partq[threadIdx.x] = aq;
  __syncthreads();
  if (threadIdx.x < 64) {
    const int t4 = threadIdx.x * 4;
    dot_z[rowbase + threadIdx.x] = partz[t4] + partz[t4 + 1] + partz[t4 + 2] + partz[t4 + 3];
    if (sumsq)
      sumsq[rowbase + threadIdx.x] = partq[t4] + partq[t4 + 1] + partq[t4 + 2] + partq[t4 + 3];
  }
}

// ---------------------------------------------------------------------------
// K2: out = (mq @ W_mem) / (dot_z + eps) + H.   M=16384, N=2048, K=384.
// Block tile 64x128. Both operand slabs are CONTIGUOUS in memory, so they are
// staged into LDS with async global->LDS DMA (ASYNCcnt), then all 48 WMMAs
// read fragments from LDS. 144KB LDS/block -> 2 blocks per WGP.
// ---------------------------------------------------------------------------
__global__ __launch_bounds__(256) void out_kernel(
    const bf16_t* __restrict__ mq_bf,  // [16384][384]
    const bf16_t* __restrict__ WmT,    // [2048][384]
    const float*  __restrict__ denom,  // [16384]
    const float*  __restrict__ H,      // [16384][2048]
    float* __restrict__ out) {         // [16384][2048]
  __shared__ __align__(64) bf16_t As[64 * 384];    // 48 KB
  __shared__ __align__(64) bf16_t Bs[128 * 384];   // 96 KB

  const int rowbase = blockIdx.x * 64;
  const int colbase = blockIdx.y * 128;
  const bf16_t* gA = mq_bf + (size_t)rowbase * 384;   // contiguous 48 KB
  const bf16_t* gB = WmT   + (size_t)colbase * 384;   // contiguous 96 KB

#if HAVE_ASYNC_LDS
#pragma unroll
  for (int i = 0; i < 12; ++i) {     // A: 3072 x 16B chunks / 256 threads
    const int c = (threadIdx.x + i * 256) * 8;
    __builtin_amdgcn_global_load_async_to_lds_b128(
        (g_int4*)(gA + c), (l_int4*)(As + c), 0, 0);
  }
#pragma unroll
  for (int i = 0; i < 24; ++i) {     // B: 6144 x 16B chunks / 256 threads
    const int c = (threadIdx.x + i * 256) * 8;
    __builtin_amdgcn_global_load_async_to_lds_b128(
        (g_int4*)(gB + c), (l_int4*)(Bs + c), 0, 0);
  }
  wait_async_lds();
#else
#pragma unroll
  for (int i = 0; i < 12; ++i) {
    const int c = (threadIdx.x + i * 256) * 8;
    *(v8bf*)(As + c) = *(const v8bf*)(gA + c);
  }
#pragma unroll
  for (int i = 0; i < 24; ++i) {
    const int c = (threadIdx.x + i * 256) * 8;
    *(v8bf*)(Bs + c) = *(const v8bf*)(gB + c);
  }
#endif
  __syncthreads();

  const int wave  = threadIdx.x >> 5;
  const int mtile = wave & 3;
  const int nhalf = wave >> 2;

  v8f c[4] = {};
#pragma unroll
  for (int k0 = 0; k0 < 384; k0 += 32) {
    v16bf a = load_a_bf16(As, 384, mtile * 16, k0);
#pragma unroll
    for (int j = 0; j < 4; ++j) {
      v16bf b = load_b_bf16(Bs, 384, nhalf * 64 + j * 16, k0);
      c[j] = WMMA_BF16(a, b, c[j]);
    }
  }
  const int lane = threadIdx.x & 31;
  const int row0 = rowbase + mtile * 16 + (lane >> 4) * 8;
  const int col0 = colbase + nhalf * 64 + (lane & 15);
#pragma unroll
  for (int v = 0; v < 8; ++v) {
    const int row = row0 + v;
    const float inv = 1.0f / (denom[row] + EPSV);
#pragma unroll
    for (int j = 0; j < 4; ++j) {
      const size_t idx = (size_t)row * 2048 + col0 + j * 16;
      out[idx] = c[j][v] * inv + H[idx];
    }
  }
}

// ---------------------------------------------------------------------------
// K5a: new_mv = mem_tokens @ Wmv^T.  M=128, N=2048, K=2048. fp32 A converted.
// ---------------------------------------------------------------------------
__global__ __launch_bounds__(256) void newmv_kernel(
    const float*  __restrict__ mem,   // [128][2048] (= out rows 16256..)
    const bf16_t* __restrict__ Wv,    // [2048][2048] bf16, row t contiguous over d
    float* __restrict__ newmv) {      // [128][2048]
  const int rowbase = blockIdx.x * 64;
  const int colbase = blockIdx.y * 128;
  const int wave  = threadIdx.x >> 5;
  const int mtile = wave & 3;
  const int nhalf = wave >> 2;
  const float* A = mem + (size_t)rowbase * 2048;

  v8f c[4] = {};
#pragma unroll 4
  for (int k0 = 0; k0 < 2048; k0 += 32) {
    v16bf a = load_a_f32(A, 2048, mtile * 16, k0);
#pragma unroll
    for (int j = 0; j < 4; ++j) {
      v16bf b = load_b_bf16(Wv, 2048, colbase + nhalf * 64 + j * 16, k0);
      c[j] = WMMA_BF16(a, b, c[j]);
    }
  }
  const int lane = threadIdx.x & 31;
  const int row0 = rowbase + mtile * 16 + (lane >> 4) * 8;
  const int col0 = colbase + nhalf * 64 + (lane & 15);
#pragma unroll
  for (int v = 0; v < 8; ++v)
#pragma unroll
    for (int j = 0; j < 4; ++j)
      newmv[(size_t)(row0 + v) * 2048 + col0 + j * 16] = c[j][v];
}

// ---------------------------------------------------------------------------
// K5b: num2 = mk @ W_mem; mv_scaled = (new_mv - num2/(d2+eps)) * mb
//      stored TRANSPOSED bf16: mvT[t][token]  (B-operand of K6).
// M=128, N=2048, K=384.
// ---------------------------------------------------------------------------
__global__ __launch_bounds__(256) void mv_kernel(
    const bf16_t* __restrict__ mk_bf,  // [128][384]
    const bf16_t* __restrict__ WmT,    // [2048][384]
    const float*  __restrict__ newmv,  // [128][2048]
    const float*  __restrict__ d2,     // [128] = z.mk (no eps)
    const float*  __restrict__ mb,     // [128]
    bf16_t* __restrict__ mvT) {        // [2048][128]
  const int rowbase = blockIdx.x * 64;
  const int colbase = blockIdx.y * 128;
  const int wave  = threadIdx.x >> 5;
  const int mtile = wave & 3;
  const int nhalf = wave >> 2;
  const bf16_t* A = mk_bf + (size_t)rowbase * 384;

  v8f c[4] = {};
#pragma unroll
  for (int k0 = 0; k0 < 384; k0 += 32) {
    v16bf a = load_a_bf16(A, 384, mtile * 16, k0);
#pragma unroll
    for (int j = 0; j < 4; ++j) {
      v16bf b = load_b_bf16(WmT, 384, colbase + nhalf * 64 + j * 16, k0);
      c[j] = WMMA_BF16(a, b, c[j]);
    }
  }
  const int lane = threadIdx.x & 31;
  const int row0 = rowbase + mtile * 16 + (lane >> 4) * 8;
  const int col0 = colbase + nhalf * 64 + (lane & 15);
#pragma unroll
  for (int v = 0; v < 8; ++v) {
    const int row = row0 + v;                        // token index
    const float inv = 1.0f / (d2[row] + EPSV);
    const float g   = mb[row];
#pragma unroll
    for (int j = 0; j < 4; ++j) {
      const int col = col0 + j * 16;
      const float val = (newmv[(size_t)row * 2048 + col] - c[j][v] * inv) * g;
      mvT[(size_t)col * 128 + row] = (bf16_t)val;
    }
  }
}

// ---------------------------------------------------------------------------
// K4: mb[t] = sigmoid(mem[t,:] . Wmb_w + b). One block per token.
// ---------------------------------------------------------------------------
__global__ __launch_bounds__(256) void mb_kernel(
    const float* __restrict__ mem,  // [128][2048]
    const float* __restrict__ Wb,   // [2048]
    const float* __restrict__ bb,   // [1]
    float* __restrict__ mb) {       // [128]
  __shared__ float part[256];
  const int t = blockIdx.x;
  float s = 0.f;
  for (int d = threadIdx.x; d < 2048; d += 256) s += mem[(size_t)t * 2048 + d] * Wb[d];
  part[threadIdx.x] = s;
  __syncthreads();
  for (int off = 128; off > 0; off >>= 1) {
    if (threadIdx.x < off) part[threadIdx.x] += part[threadIdx.x + off];
    __syncthreads();
  }
  if (threadIdx.x == 0) mb[t] = 1.f / (1.f + expf(-(part[0] + bb[0])));
}

// ---------------------------------------------------------------------------
// K6: W_mem_new = W_mem + mk^T @ mv_scaled.  M=384, N=2048, K=128 (4 iters).
// A = mkT [384][128] (K contiguous), B = mvT [2048][128] (K contiguous).
// ---------------------------------------------------------------------------
__global__ __launch_bounds__(256) void assoc_kernel(
    const bf16_t* __restrict__ mkT,   // [384][128]
    const bf16_t* __restrict__ mvT,   // [2048][128]
    const float*  __restrict__ Wmem,  // [384][2048] fp32
    float* __restrict__ Wout) {       // [384][2048]
  const int rowbase = blockIdx.x * 64;
  const int colbase = blockIdx.y * 128;
  const int wave  = threadIdx.x >> 5;
  const int mtile = wave & 3;
  const int nhalf = wave >> 2;
  const bf16_t* A = mkT + (size_t)rowbase * 128;

  v8f c[4] = {};
#pragma unroll
  for (int k0 = 0; k0 < 128; k0 += 32) {
    v16bf a = load_a_bf16(A, 128, mtile * 16, k0);
#pragma unroll
    for (int j = 0; j < 4; ++j) {
      v16bf b = load_b_bf16(mvT, 128, colbase + nhalf * 64 + j * 16, k0);
      c[j] = WMMA_BF16(a, b, c[j]);
    }
  }
  const int lane = threadIdx.x & 31;
  const int row0 = rowbase + mtile * 16 + (lane >> 4) * 8;
  const int col0 = colbase + nhalf * 64 + (lane & 15);
#pragma unroll
  for (int v = 0; v < 8; ++v)
#pragma unroll
    for (int j = 0; j < 4; ++j) {
      const size_t idx = (size_t)(row0 + v) * 2048 + col0 + j * 16;
      Wout[idx] = Wmem[idx] + c[j][v];
    }
}

// ---------------------------------------------------------------------------
// K7: z_new[j] = z[j] + sum_t clip(1 - (d2[t]+eps)/(sq[t]+eps), 0, 1) * mk[t][j]
// ---------------------------------------------------------------------------
__global__ __launch_bounds__(384) void znew_kernel(
    const float* __restrict__ mk_f32,  // [128][384]
    const float* __restrict__ d2,      // [128]
    const float* __restrict__ sq,      // [128]
    const float* __restrict__ z,       // [384]
    float* __restrict__ zout) {        // [384]
  __shared__ float coef[128];
  if (threadIdx.x < 128) {
    float c = 1.f - (d2[threadIdx.x] + EPSV) / (sq[threadIdx.x] + EPSV);
    coef[threadIdx.x] = fminf(fmaxf(c, 0.f), 1.f);
  }
  __syncthreads();
  const int j = threadIdx.x;
  if (j < 384) {
    float s = 0.f;
    for (int t = 0; t < 128; ++t) s += coef[t] * mk_f32[(size_t)t * 384 + j];
    zout[j] = z[j] + s;
  }
}

// ---------------------------------------------------------------------------
extern "C" void kernel_launch(void* const* d_in, const int* in_sizes, int n_in,
                              void* d_out, int out_size, void* d_ws, size_t ws_size,
                              hipStream_t stream) {
  const float* H    = (const float*)d_in[0];   // [1][16384][2048]
  const float* Wmq  = (const float*)d_in[1];   // [64][2048]
  const float* Wmk  = (const float*)d_in[2];   // [64][2048]
  const float* Wmv  = (const float*)d_in[3];   // [2048][2048]
  const float* Wmbw = (const float*)d_in[4];   // [1][2048]
  const float* Wmbb = (const float*)d_in[5];   // [1]
  const float* Wmem = (const float*)d_in[6];   // [1][384][2048]
  const float* z    = (const float*)d_in[7];   // [1][384]

  float* out  = (float*)d_out;                       // 16384*2048
  float* Wout = out + (size_t)16384 * 2048;          // 384*2048
  float* zout = Wout + (size_t)384 * 2048;           // 384

  char* ws = (char*)d_ws;
  size_t off = 0;
  auto alloc = [&](size_t bytes) -> char* {
    char* p = ws + off;
    off += (bytes + 255) & ~(size_t)255;
    return p;
  };
  bf16_t* Wmq_bf  = (bf16_t*)alloc((size_t)64 * 2048 * 2);
  bf16_t* Wmk_bf  = (bf16_t*)alloc((size_t)64 * 2048 * 2);
  bf16_t* Wmv_bf  = (bf16_t*)alloc((size_t)2048 * 2048 * 2);
  bf16_t* WmT_bf  = (bf16_t*)alloc((size_t)2048 * 384 * 2);   // W_mem transposed
  bf16_t* mq_bf   = (bf16_t*)alloc((size_t)16384 * 384 * 2);
  float*  denom   = (float*) alloc((size_t)16384 * 4);
  bf16_t* mk_bf   = (bf16_t*)alloc((size_t)128 * 384 * 2);
  bf16_t* mkT_bf  = (bf16_t*)alloc((size_t)384 * 128 * 2);
  float*  mk_f32  = (float*) alloc((size_t)128 * 384 * 4);
  float*  d2      = (float*) alloc((size_t)128 * 4);
  float*  sq      = (float*) alloc((size_t)128 * 4);
  float*  mb      = (float*) alloc((size_t)128 * 4);
  float*  newmv   = (float*) alloc((size_t)128 * 2048 * 4);
  bf16_t* mvT_bf  = (bf16_t*)alloc((size_t)2048 * 128 * 2);
  (void)ws_size; (void)in_sizes; (void)n_in; (void)out_size;

  // K0: weight conversions
  cvt_kernel<<<(64 * 2048 + 255) / 256, 256, 0, stream>>>(Wmq, Wmq_bf, 64 * 2048);
  cvt_kernel<<<(64 * 2048 + 255) / 256, 256, 0, stream>>>(Wmk, Wmk_bf, 64 * 2048);
  cvt_kernel<<<(2048 * 2048 + 255) / 256, 256, 0, stream>>>(Wmv, Wmv_bf, 2048 * 2048);
  transpose_cvt_kernel<<<(384 * 2048 + 255) / 256, 256, 0, stream>>>(Wmem, WmT_bf);

  // K1: mq = dpfp(H @ Wmq^T), denom = mq.z   (16384 rows, 64 rows/block)
  mq_kernel<<<256, 256, 0, stream>>>(H, Wmq_bf, z, mq_bf,
                                     (bf16_t*)nullptr, (float*)nullptr,
                                     denom, (float*)nullptr, 0);

  // K2: out = (mq @ W_mem)/(denom+eps) + H  (async-LDS staged WMMA GEMM)
  out_kernel<<<dim3(256, 16), 256, 0, stream>>>(mq_bf, WmT_bf, denom, H, out);

  const float* mem = out + (size_t)(16384 - 128) * 2048;

  // K3: mk = dpfp(mem @ Wmk^T), d2 = z.mk, sq = sum(mk^2)   (128 rows)
  mq_kernel<<<2, 256, 0, stream>>>(mem, Wmk_bf, z, mk_bf,
                                   mkT_bf, mk_f32, d2, sq, 128);

  // K4: mb = sigmoid(mem . Wmb_w + b)
  mb_kernel<<<128, 256, 0, stream>>>(mem, Wmbw, Wmbb, mb);

  // K5a: new_mv = mem @ Wmv^T
  newmv_kernel<<<dim3(2, 16), 256, 0, stream>>>(mem, Wmv_bf, newmv);

  // K5b: mvT = ((new_mv - (mk @ W_mem)/(d2+eps)) * mb)^T  (bf16)
  mv_kernel<<<dim3(2, 16), 256, 0, stream>>>(mk_bf, WmT_bf, newmv, d2, mb, mvT_bf);

  // K6: W_mem_new = W_mem + mk^T @ mv_scaled
  assoc_kernel<<<dim3(6, 16), 256, 0, stream>>>(mkT_bf, mvT_bf, Wmem, Wout);

  // K7: z_new
  znew_kernel<<<1, 384, 0, stream>>>(mk_f32, d2, sq, z, zout);
}
